// NFG4img_35364760715528
// MI455X (gfx1250) — matrix operand order repaired and checked by reading, not compiled, not run on hardware
//
#include <hip/hip_runtime.h>
#include <hip/hip_bf16.h>

// MI455X / gfx1250, wave32. GEMMs use v_wmma_f32_16x16x32_f16 (f16 in, f32 acc).
// Workspace layout (bytes), total ~71.6 MB:
//   [0            ) v16   : 131072*256 f16 = 67108864   (v = inps@wv + b, unscaled)
//   [67108864     ) g     : 131072*8  f32  = 4194304    (relu gate)
//   [71303168     ) wvF   : 65536 f16      = 131072     (wv_w WMMA B-fragments)
//   [71434240     ) wfF   : 65536 f16      = 131072     (wf_w WMMA B-fragments)
//   [71565312     ) gkF   : 4096  f16      = 8192       (wgactk padded N->16 fragments)
//   [71573504     ) s0    : 128 f32
//   [71574016     ) qv    : 8 f32
//   [71574048     ) scale : 256 f32        (per-batch,head act scale)

typedef __attribute__((ext_vector_type(16))) _Float16 v16h;
typedef __attribute__((ext_vector_type(4)))  _Float16 v4h;
typedef __attribute__((ext_vector_type(8)))  float    v8f;

#define OFF_V16   0
#define OFF_G     67108864
#define OFF_WVF   71303168
#define OFF_WFF   71434240
#define OFF_GKF   71565312
#define OFF_S0    71573504
#define OFF_QV    71574016
#define OFF_SC    71574048

// ---------------------------------------------------------------------------
// Pre-swizzle weight matrices into per-lane WMMA B-fragment layout.
// B(k,n) for 16x16x32 f16 tile: lane = (n%16) + 16*((k%16)/8), half = k%8 + 8*(k/16).
// Fragment storage: [kb(0..7)][nt][lane(0..31)][h(0..15)] contiguous halves.
__global__ void nfg_kw_frag(const float* __restrict__ wv, const float* __restrict__ wf,
                            const float* __restrict__ gk,
                            _Float16* __restrict__ wvF, _Float16* __restrict__ wfF,
                            _Float16* __restrict__ gkF) {
    int nthr = gridDim.x * blockDim.x;
    int tid  = blockIdx.x * blockDim.x + threadIdx.x;
    for (int d = tid; d < 8 * 16 * 32 * 16; d += nthr) {
        int h = d & 15, lane = (d >> 4) & 31, nt = (d >> 9) & 15, kb = d >> 13;
        int k = kb * 32 + ((lane >> 4) << 3) + (h & 7) + ((h >> 3) << 4);
        int n = nt * 16 + (lane & 15);
        wvF[d] = (_Float16)wv[k * 256 + n];
        wfF[d] = (_Float16)wf[k * 256 + n];
    }
    for (int d = tid; d < 8 * 32 * 16; d += nthr) {
        int h = d & 15, lane = (d >> 4) & 31, kb = d >> 9;
        int k = kb * 32 + ((lane >> 4) << 3) + (h & 7) + ((h >> 3) << 4);
        int n = lane & 15;
        gkF[d] = (n < 8) ? (_Float16)gk[k * 8 + n] : (_Float16)0.0f;
    }
}

// ---------------------------------------------------------------------------
// s0[h][j] = softmax_j( kq[0][128+h*16+j] + kq[j][h*16] ); qv[h] = actq[14][h]
__global__ void nfg_setup(const float* __restrict__ ne, const float* __restrict__ wkq,
                          const float* __restrict__ wkqb, const float* __restrict__ wgq,
                          const float* __restrict__ wgqb,
                          float* __restrict__ s0g, float* __restrict__ qvg) {
    __shared__ float kqs[16 * 256];
    int c = threadIdx.x;                      // 256 threads, one output column each
    float s[16];
    #pragma unroll
    for (int n = 0; n < 16; ++n) s[n] = wkqb[c];
    for (int d = 0; d < 128; ++d) {
        float w = wkq[d * 256 + c];
        #pragma unroll
        for (int n = 0; n < 16; ++n) s[n] += ne[n * 128 + d] * w;
    }
    #pragma unroll
    for (int n = 0; n < 16; ++n) kqs[n * 256 + c] = s[n];
    __syncthreads();
    if (c < 8) {
        int h = c;
        float L[16], mx = -1e30f;
        #pragma unroll
        for (int j = 0; j < 16; ++j) {
            L[j] = kqs[128 + h * 16 + j] + kqs[j * 256 + h * 16];
            mx = fmaxf(mx, L[j]);
        }
        float sum = 0.0f;
        #pragma unroll
        for (int j = 0; j < 16; ++j) { L[j] = expf(L[j] - mx); sum += L[j]; }
        float inv = 1.0f / sum;
        #pragma unroll
        for (int j = 0; j < 16; ++j) s0g[h * 16 + j] = L[j] * inv;
        float q = wgqb[h];
        for (int d = 0; d < 128; ++d) q += ne[14 * 128 + d] * wgq[d * 8 + h];
        qvg[h] = q;
    }
}

// ---------------------------------------------------------------------------
// Fused: v = inps@wv + b (f16 out) and g = relu(qv + inps@wgactk + b).
// Block = 32 pixels x 256 channels; 8 waves = 2(M) x 4(N); K=256 in 8 steps.
__global__ void __launch_bounds__(256) nfg_vact(const float* __restrict__ inps,
    const _Float16* __restrict__ wvF, const _Float16* __restrict__ gkF,
    const float* __restrict__ wvb, const float* __restrict__ gkb,
    const float* __restrict__ qvg,
    _Float16* __restrict__ v16, float* __restrict__ g) {
    __shared__ _Float16 sA[2 * 8 * 32 * 16];   // A fragments, 16 KB
    const long p0 = (long)blockIdx.x * 32;
    const int tid = threadIdx.x;

    // Stage 32x256 f32 tile -> f16 A-fragment layout in LDS.
    // A(m,k): lane = m%16 + 16*((k%16)/8), half = k%8 + 8*(k/16).
    // 4-aligned channel quads land in 4 consecutive LDS halves -> b64 stores.
    {
        const int q = tid & 63, mg = tid >> 6;    // channel quad, row group
        const int k = 4 * q;
        const int kb = k >> 5, kk = k & 31;
        const int laneS = (((kk & 15) >> 3) << 4);
        const int hb = (kk & 7) + ((kk >> 4) << 3);
        #pragma unroll
        for (int i = 0; i < 8; ++i) {
            int m = mg * 8 + i;
            float4 f = *(const float4*)(inps + (p0 + m) * 256 + k);
            v4h hv = { (_Float16)f.x, (_Float16)f.y, (_Float16)f.z, (_Float16)f.w };
            *(v4h*)(&sA[(((m >> 4) * 8 + kb) * 32 + ((m & 15) + laneS)) * 16 + hb]) = hv;
        }
    }
    __syncthreads();

    const int wid = tid >> 5, lane = tid & 31;
    const int mt = wid >> 2, ng = wid & 3;
    const v16h* aF = (const v16h*)sA;
    const v16h* bF = (const v16h*)wvF;
    const v16h* gF = (const v16h*)gkF;
    v8f acc[4] = {};
    v8f accA = {};
    for (int kb = 0; kb < 8; ++kb) {
        v16h A = aF[(mt * 8 + kb) * 32 + lane];
        #pragma unroll
        for (int t = 0; t < 4; ++t) {
            v16h B = bF[(kb * 16 + (ng * 4 + t)) * 32 + lane];
            acc[t] = __builtin_amdgcn_wmma_f32_16x16x32_f16(
                false, A, false, B, (short)0, acc[t], false, false);
        }
        if (ng == 0) {   // wave-uniform branch: EXEC stays all-ones inside
            v16h Bg = gF[kb * 32 + lane];
            accA = __builtin_amdgcn_wmma_f32_16x16x32_f16(
                false, A, false, Bg, (short)0, accA, false, false);
        }
    }
    // C/D layout: VGPR r holds M = r + 8*(lane>=16), N = lane%16.
    const int rowHalf = (lane >> 4) << 3;
    #pragma unroll
    for (int t = 0; t < 4; ++t) {
        int nc = ng * 64 + t * 16 + (lane & 15);
        float bias = wvb[nc];
        #pragma unroll
        for (int r = 0; r < 8; ++r) {
            long p = p0 + mt * 16 + r + rowHalf;
            v16[p * 256 + nc] = (_Float16)(acc[t][r] + bias);
        }
    }
    if (ng == 0) {
        int h = lane & 15;
        if (h < 8) {
            float base = qvg[h] + gkb[h];
            #pragma unroll
            for (int r = 0; r < 8; ++r) {
                long p = p0 + mt * 16 + r + rowHalf;
                g[p * 8 + h] = fmaxf(accA[r] + base, 0.0f);
            }
        }
    }
}

// ---------------------------------------------------------------------------
// scale[b][h] = 4096 / (sum_{y,x} g + 1e-6), deterministic tree reduce.
__global__ void nfg_denom(const float* __restrict__ g, float* __restrict__ scaleB) {
    __shared__ float sh[256];
    int b = blockIdx.x, t = threadIdx.x;
    int h = t & 7, row = t >> 3;
    const float* gb = g + (long)b * 4096 * 8;
    float s = 0.0f;
    for (int p = row; p < 4096; p += 32) s += gb[p * 8 + h];
    sh[t] = s;
    __syncthreads();
    if (t < 8) {
        float tot = 0.0f;
        #pragma unroll
        for (int r = 0; r < 32; ++r) tot += sh[r * 8 + t];
        scaleB[b * 8 + t] = 4096.0f / (tot + 1e-6f);
    }
}

// ---------------------------------------------------------------------------
// Patch gather (fold act-scale + s0 weight) + final 256x256 GEMM + bias.
// Block = 32 output pixels (same batch, same Y row) x 256 channels.
// Gather phase: thread owns a 4-channel quad (same head) x 8 output rows,
// so each v read is a coalesced b64 and each LDS write a b64.
__global__ void __launch_bounds__(256) nfg_out(const _Float16* __restrict__ v16,
    const float* __restrict__ g, const float* __restrict__ s0g,
    const float* __restrict__ scaleB, const _Float16* __restrict__ wfF,
    const float* __restrict__ wfb, float* __restrict__ out) {
    __shared__ _Float16 sA[2 * 8 * 32 * 16];
    const int op0 = blockIdx.x * 32;
    const int b = op0 >> 10;                  // 32x32 = 1024 outputs per batch
    const int tid = threadIdx.x;

    const int q = tid & 63, mg = tid >> 6;    // channel quad, row group
    const int c = 4 * q;
    const int h = c >> 5;                     // whole quad in one head
    const float sc = scaleB[b * 8 + h];
    float s0v[16];
    #pragma unroll
    for (int n = 0; n < 16; ++n) s0v[n] = s0g[h * 16 + n] * sc;

    const int kb = c >> 5, kk = c & 31;
    const int lA = (((kk & 15) >> 3) << 4);
    const int hA = (kk & 7) + ((kk >> 4) << 3);

    #pragma unroll
    for (int i = 0; i < 8; ++i) {
        int m = mg * 8 + i;
        int op = op0 + m;
        int Y = (op >> 5) & 31, X = op & 31;
        float a0 = 0.0f, a1 = 0.0f, a2 = 0.0f, a3 = 0.0f;
        #pragma unroll
        for (int n = 0; n < 16; ++n) {
            int ys = 2 * Y - 1 + (n >> 2);    // pad_h//2 == 1
            int xs = 2 * X - 1 + (n & 3);
            if ((unsigned)ys < 64u && (unsigned)xs < 64u) {
                long src = (long)b * 4096 + ys * 64 + xs;
                float w = s0v[n] * g[src * 8 + h];
                v4h vv = *(const v4h*)(v16 + src * 256 + c);
                a0 += w * (float)vv.x;
                a1 += w * (float)vv.y;
                a2 += w * (float)vv.z;
                a3 += w * (float)vv.w;
            }
        }
        v4h hv = { (_Float16)a0, (_Float16)a1, (_Float16)a2, (_Float16)a3 };
        *(v4h*)(&sA[(((m >> 4) * 8 + kb) * 32 + ((m & 15) + lA)) * 16 + hA]) = hv;
    }
    __syncthreads();

    const int wid = tid >> 5, lane = tid & 31;
    const int mt = wid >> 2, ng = wid & 3;
    const v16h* aF = (const v16h*)sA;
    const v16h* bF = (const v16h*)wfF;
    v8f accD[4] = {};
    for (int k2 = 0; k2 < 8; ++k2) {
        v16h A = aF[(mt * 8 + k2) * 32 + lane];
        #pragma unroll
        for (int t = 0; t < 4; ++t) {
            v16h B = bF[(k2 * 16 + (ng * 4 + t)) * 32 + lane];
            accD[t] = __builtin_amdgcn_wmma_f32_16x16x32_f16(
                false, A, false, B, (short)0, accD[t], false, false);
        }
    }
    const int rowHalf = (lane >> 4) << 3;
    #pragma unroll
    for (int t = 0; t < 4; ++t) {
        int nc = ng * 64 + t * 16 + (lane & 15);
        float bias = wfb[nc];
        #pragma unroll
        for (int r = 0; r < 8; ++r) {
            long op = op0 + mt * 16 + r + rowHalf;
            out[op * 256 + nc] = accD[t][r] + bias;
        }
    }
}

// ---------------------------------------------------------------------------
extern "C" void kernel_launch(void* const* d_in, const int* in_sizes, int n_in,
                              void* d_out, int out_size, void* d_ws, size_t ws_size,
                              hipStream_t stream) {
    (void)in_sizes; (void)n_in; (void)out_size; (void)ws_size;
    const float* inps = (const float*)d_in[0];
    const float* ne   = (const float*)d_in[1];
    const float* wkq  = (const float*)d_in[2];
    const float* wkqb = (const float*)d_in[3];
    const float* wv   = (const float*)d_in[4];
    const float* wvb  = (const float*)d_in[5];
    const float* wgq  = (const float*)d_in[6];
    const float* wgqb = (const float*)d_in[7];
    const float* wgk  = (const float*)d_in[8];
    const float* wgkb = (const float*)d_in[9];
    const float* wf   = (const float*)d_in[10];
    const float* wfb  = (const float*)d_in[11];
    float* out = (float*)d_out;

    char* ws = (char*)d_ws;
    _Float16* v16   = (_Float16*)(ws + OFF_V16);
    float*    g     = (float*)(ws + OFF_G);
    _Float16* wvF   = (_Float16*)(ws + OFF_WVF);
    _Float16* wfF   = (_Float16*)(ws + OFF_WFF);
    _Float16* gkF   = (_Float16*)(ws + OFF_GKF);
    float*    s0g   = (float*)(ws + OFF_S0);
    float*    qvg   = (float*)(ws + OFF_QV);
    float*    sclB  = (float*)(ws + OFF_SC);

    nfg_kw_frag<<<64, 256, 0, stream>>>(wv, wf, wgk, wvF, wfF, gkF);
    nfg_setup<<<1, 256, 0, stream>>>(ne, wkq, wkqb, wgq, wgqb, s0g, qvg);
    nfg_vact<<<4096, 256, 0, stream>>>(inps, wvF, gkF, wvb, wgkb, qvg, v16, g);
    nfg_denom<<<32, 256, 0, stream>>>(g, sclB);
    nfg_out<<<1024, 256, 0, stream>>>(v16, g, s0g, sclB, wfF, wfb, out);
}
